// LSTM_47270410060255
// MI455X (gfx1250) — compile-verified, matrix operand order checked by reading, not run on hardware
//
#include <hip/hip_runtime.h>
#include <hip/hip_bf16.h>

// Persistent 2-layer LSTM (B=64, T=256, D=512, H=1024) for MI455X / gfx1250.
// - bf16 WMMA (v_wmma_f32_16x16x32_bf16), fp32 accumulation, fp32 cell state.
// - One persistent kernel runs all 256 timesteps; grid barrier between phases
//   (atomic counter + s_sleep spin) removes ~1000 launch latencies from the
//   serial recurrence.
// - Each wave owns 1 M-tile x the 4 gate slices (j, j+H, j+2H, j+3H) of one
//   j-tile, so the LSTM pointwise (sigmoid/tanh, c update, bf16 h write)
//   happens in-register right after the K-reduction: no gates buffer.
// - h0/h1 double-buffered; c owner-only.
// - K loops pinned to unroll 2: keeps VGPR pressure < 256 (round-2 full
//   unroll spilled to scratch inside the WMMA loop).

#define D_IN    512
#define HDIM    1024
#define OUT_DIM 512
#define BATCH   64
#define TSTEPS  256
#define G4H     (4 * HDIM)
#define NBLOCKS 32          // 32 blocks x 8 waves = 256 waves (4 mt x 64 jt)

typedef __attribute__((ext_vector_type(16))) __bf16 v16bf;
typedef __attribute__((ext_vector_type(8)))  float  v8f;

__device__ __forceinline__ __bf16 f2bf(float f) {
  unsigned int u = __float_as_uint(f);
  u += 0x7FFFu + ((u >> 16) & 1u);          // round-to-nearest-even
  unsigned short s = (unsigned short)(u >> 16);
  __bf16 r;
  __builtin_memcpy(&r, &s, sizeof(r));
  return r;
}

__device__ __forceinline__ float sigmoidf_fast(float x) {
  return 1.0f / (1.0f + __expf(-x));
}
__device__ __forceinline__ float tanhf_fast(float x) {
  return 1.0f - 2.0f / (1.0f + __expf(2.0f * x));
}

// ---------------------------------------------------------------- conversion
__global__ void __launch_bounds__(256)
f32_to_bf16_kernel(const float* __restrict__ src, __bf16* __restrict__ dst, size_t n) {
  size_t i = (size_t)blockIdx.x * blockDim.x + threadIdx.x;
  size_t stride = (size_t)gridDim.x * blockDim.x;
  for (; i < n; i += stride) dst[i] = f2bf(src[i]);
}

__global__ void __launch_bounds__(256)
zero_u32_kernel(unsigned int* __restrict__ p, size_t nwords) {
  size_t i = (size_t)blockIdx.x * blockDim.x + threadIdx.x;
  size_t stride = (size_t)gridDim.x * blockDim.x;
  for (; i < nwords; i += stride) p[i] = 0u;
}

// ------------------------------------------------------------- grid barrier
// Monotonic counter barrier: each block contributes +1 per barrier; expected
// value is barrier_index * NBLOCKS. Counter zeroed by host before launch.
__device__ __forceinline__ void grid_barrier(unsigned int* ctr, unsigned int expected) {
  __threadfence();                      // release my writes to device scope
  __syncthreads();
  if (threadIdx.x == 0) {
    __hip_atomic_fetch_add(ctr, 1u, __ATOMIC_ACQ_REL, __HIP_MEMORY_SCOPE_AGENT);
    while (__hip_atomic_load(ctr, __ATOMIC_ACQUIRE, __HIP_MEMORY_SCOPE_AGENT) < expected) {
      __builtin_amdgcn_s_sleep(1);
    }
  }
  __syncthreads();
  __threadfence();                      // acquire: don't read stale lines
}

// -------------------------------------------------- fused dual-GEMM + cell
// For this wave's (mt, jt): computes the 4 gate pre-activations
//   g[m, j + g*H] = A1[m,:K1]@W1[n_g,:K1] + A2[m,:K2]@W2[n_g,:K2] + b1 + b2
// then applies the LSTM cell in-register: c_new, h_new -> c[], h_out[] (bf16).
// Fragment layout used by this code (contiguous k = 16h + s, same order for A and B):
//   A: lane L holds row (L&15), 32B K-slab at offset (L>>4)*16
//   B: lane L holds W row n_g (= B column), same K-slab split
//   C/D: VGPR r -> M = r + 8*(L>>4), N-col = (L&15)
__device__ __forceinline__ void lstm_layer_tile(
    const __bf16* __restrict__ A1, long lda1, int K1, const __bf16* __restrict__ W1,
    const __bf16* __restrict__ A2, int K2, const __bf16* __restrict__ W2,
    const float* __restrict__ b1, const float* __restrict__ b2,
    float* __restrict__ c, __bf16* __restrict__ h_out,
    int mt, int jt, int lane) {
  const int kh   = (lane >> 4) << 4;
  const int col  = lane & 15;
  const int rowA = (mt << 4) + col;

  int n[4];
  v8f acc[4];
  #pragma unroll
  for (int g = 0; g < 4; ++g) {
    n[g] = g * HDIM + (jt << 4) + col;
    const float b = b1[n[g]] + b2[n[g]];
    #pragma unroll
    for (int r = 0; r < 8; ++r) acc[g][r] = b;
  }

  const __bf16* a1p = A1 + (long)rowA * lda1 + kh;
  #pragma unroll 2
  for (int k0 = 0; k0 < K1; k0 += 32) {
    v16bf a = *(const v16bf*)(a1p + k0);
    __builtin_prefetch(W1 + (long)n[0] * K1 + k0 + 32 + kh, 0, 1);
    #pragma unroll
    for (int g = 0; g < 4; ++g) {
      v16bf bf = *(const v16bf*)(W1 + (long)n[g] * K1 + k0 + kh);
      acc[g] = __builtin_amdgcn_wmma_f32_16x16x32_bf16(
          false, a, false, bf, (short)0, acc[g], false, false);
    }
  }
  const __bf16* a2p = A2 + (long)rowA * K2 + kh;
  #pragma unroll 2
  for (int k0 = 0; k0 < K2; k0 += 32) {
    v16bf a = *(const v16bf*)(a2p + k0);
    __builtin_prefetch(W2 + (long)n[0] * K2 + k0 + 32 + kh, 0, 1);
    #pragma unroll
    for (int g = 0; g < 4; ++g) {
      v16bf bf = *(const v16bf*)(W2 + (long)n[g] * K2 + k0 + kh);
      acc[g] = __builtin_amdgcn_wmma_f32_16x16x32_bf16(
          false, a, false, bf, (short)0, acc[g], false, false);
    }
  }

  // In-register LSTM cell: acc[0]=i, acc[1]=f, acc[2]=g, acc[3]=o.
  const int m0 = (mt << 4) + ((lane >> 4) << 3);
  const int j  = (jt << 4) + col;
  #pragma unroll
  for (int r = 0; r < 8; ++r) {
    const long idx = (long)(m0 + r) * HDIM + j;
    const float cn = c[idx] * sigmoidf_fast(acc[1][r])
                   + sigmoidf_fast(acc[0][r]) * tanhf_fast(acc[2][r]);
    const float hn = sigmoidf_fast(acc[3][r]) * tanhf_fast(cn);
    c[idx]     = cn;
    h_out[idx] = f2bf(hn);
  }
}

// ----------------------------------------------------------- persistent LSTM
__global__ void __launch_bounds__(256)
persistent_lstm_kernel(const __bf16* __restrict__ xb,
                       const __bf16* __restrict__ wxh0b, const __bf16* __restrict__ whh0b,
                       const __bf16* __restrict__ wxh1b, const __bf16* __restrict__ whh1b,
                       const __bf16* __restrict__ wfcb,
                       const float* __restrict__ bxh0, const float* __restrict__ bhh0,
                       const float* __restrict__ bxh1, const float* __restrict__ bhh1,
                       const float* __restrict__ bfc,
                       float* __restrict__ c0, float* __restrict__ c1,
                       __bf16* __restrict__ h0buf, __bf16* __restrict__ h1buf,
                       float* __restrict__ out,
                       unsigned int* __restrict__ barrier_ctr) {
  const int lane = threadIdx.x & 31;
  const int wave = threadIdx.x >> 5;
  const int w    = blockIdx.x * 8 + wave;   // 0..255
  const int mt   = w >> 6;                  // 0..3
  const int jt   = w & 63;                  // 0..63
  const long BH  = (long)BATCH * HDIM;

  unsigned int bar = 0;
  #pragma unroll 1
  for (int t = 0; t < TSTEPS; ++t) {
    const __bf16* h0_cur = h0buf + (size_t)(t & 1) * BH;
    __bf16*       h0_nxt = h0buf + (size_t)((t + 1) & 1) * BH;
    const __bf16* h1_cur = h1buf + (size_t)(t & 1) * BH;
    __bf16*       h1_nxt = h1buf + (size_t)((t + 1) & 1) * BH;

    // layer 0: gates = x_t @ Wxh0^T + h0 @ Whh0^T + bxh0 + bhh0 -> c0, h0_nxt
    lstm_layer_tile(xb + (size_t)t * D_IN, (long)TSTEPS * D_IN, D_IN, wxh0b,
                    h0_cur, HDIM, whh0b, bxh0, bhh0, c0, h0_nxt, mt, jt, lane);
    grid_barrier(barrier_ctr, (++bar) * NBLOCKS);

    // layer 1: gates = h0_nxt @ Wxh1^T + h1 @ Whh1^T + bxh1 + bhh1 -> c1, h1_nxt
    lstm_layer_tile(h0_nxt, HDIM, HDIM, wxh1b,
                    h1_cur, HDIM, whh1b, bxh1, bhh1, c1, h1_nxt, mt, jt, lane);
    grid_barrier(barrier_ctr, (++bar) * NBLOCKS);
  }

  // Final FC on last h1 (buffer index TSTEPS & 1 == 0): waves 0..31,
  // each wave: 1 M-tile x 4 N-tiles of out[64, 512] = h1 @ Wfc^T + bfc.
  if (w < 32) {
    const __bf16* h1f = h1buf + (size_t)(TSTEPS & 1) * BH;
    const int fmt   = w >> 3;                 // 0..3
    const int nbase = (w & 7) << 6;           // 0,64,..,448
    const int kh    = (lane >> 4) << 4;
    const int col   = lane & 15;
    const int rowA  = (fmt << 4) + col;

    v8f acc[4];
    #pragma unroll
    for (int j = 0; j < 4; ++j) {
      const float b = bfc[nbase + (j << 4) + col];
      #pragma unroll
      for (int r = 0; r < 8; ++r) acc[j][r] = b;
    }
    const __bf16* ap = h1f + (long)rowA * HDIM + kh;
    #pragma unroll 2
    for (int k0 = 0; k0 < HDIM; k0 += 32) {
      v16bf a = *(const v16bf*)(ap + k0);
      #pragma unroll
      for (int j = 0; j < 4; ++j) {
        const int n = nbase + (j << 4) + col;
        v16bf bf = *(const v16bf*)(wfcb + (long)n * HDIM + k0 + kh);
        acc[j] = __builtin_amdgcn_wmma_f32_16x16x32_bf16(
            false, a, false, bf, (short)0, acc[j], false, false);
      }
    }
    const int m0 = (fmt << 4) + ((lane >> 4) << 3);
    #pragma unroll
    for (int j = 0; j < 4; ++j) {
      const int n = nbase + (j << 4) + col;
      #pragma unroll
      for (int r = 0; r < 8; ++r)
        out[(long)(m0 + r) * OUT_DIM + n] = acc[j][r];
    }
  }
}

// ------------------------------------------------------------------- launcher
extern "C" void kernel_launch(void* const* d_in, const int* in_sizes, int n_in,
                              void* d_out, int out_size, void* d_ws, size_t ws_size,
                              hipStream_t stream) {
  const float* x    = (const float*)d_in[0];
  const float* Wxh0 = (const float*)d_in[1];
  const float* bxh0 = (const float*)d_in[2];
  const float* Whh0 = (const float*)d_in[3];
  const float* bhh0 = (const float*)d_in[4];
  const float* Wxh1 = (const float*)d_in[5];
  const float* bxh1 = (const float*)d_in[6];
  const float* Whh1 = (const float*)d_in[7];
  const float* bhh1 = (const float*)d_in[8];
  const float* Wfc  = (const float*)d_in[9];
  const float* bfc  = (const float*)d_in[10];
  float* out = (float*)d_out;

  // -------- workspace carve-out (~46 MB), 256B-aligned slices
  char* ws = (char*)d_ws;
  size_t off = 0;
  auto take = [&](size_t bytes) -> void* {
    void* p = ws + off;
    off += (bytes + 255) & ~(size_t)255;
    return p;
  };
  __bf16* wxh0b = (__bf16*)take((size_t)G4H * D_IN * 2);
  __bf16* whh0b = (__bf16*)take((size_t)G4H * HDIM * 2);
  __bf16* wxh1b = (__bf16*)take((size_t)G4H * HDIM * 2);
  __bf16* whh1b = (__bf16*)take((size_t)G4H * HDIM * 2);
  __bf16* wfcb  = (__bf16*)take((size_t)OUT_DIM * HDIM * 2);
  __bf16* xb    = (__bf16*)take((size_t)BATCH * TSTEPS * D_IN * 2);
  float*  c0    = (float*)take((size_t)BATCH * HDIM * 4);
  float*  c1    = (float*)take((size_t)BATCH * HDIM * 4);
  __bf16* h0buf = (__bf16*)take((size_t)2 * BATCH * HDIM * 2);   // ping-pong
  __bf16* h1buf = (__bf16*)take((size_t)2 * BATCH * HDIM * 2);   // ping-pong
  unsigned int* bar = (unsigned int*)take(256);

  // -------- per-launch conversions + state/barrier init (all on stream)
  const int CG = 2048;
  f32_to_bf16_kernel<<<CG, 256, 0, stream>>>(Wxh0, wxh0b, (size_t)G4H * D_IN);
  f32_to_bf16_kernel<<<CG, 256, 0, stream>>>(Whh0, whh0b, (size_t)G4H * HDIM);
  f32_to_bf16_kernel<<<CG, 256, 0, stream>>>(Wxh1, wxh1b, (size_t)G4H * HDIM);
  f32_to_bf16_kernel<<<CG, 256, 0, stream>>>(Whh1, whh1b, (size_t)G4H * HDIM);
  f32_to_bf16_kernel<<<CG, 256, 0, stream>>>(Wfc,  wfcb,  (size_t)OUT_DIM * HDIM);
  f32_to_bf16_kernel<<<CG, 256, 0, stream>>>(x,    xb,    (size_t)BATCH * TSTEPS * D_IN);
  zero_u32_kernel<<<256, 256, 0, stream>>>((unsigned int*)c0, (size_t)BATCH * HDIM);
  zero_u32_kernel<<<256, 256, 0, stream>>>((unsigned int*)c1, (size_t)BATCH * HDIM);
  // zero only ping buffer 0 of h0/h1 (read at t=0); buffer 1 written first.
  zero_u32_kernel<<<256, 256, 0, stream>>>((unsigned int*)h0buf, (size_t)BATCH * HDIM / 2);
  zero_u32_kernel<<<256, 256, 0, stream>>>((unsigned int*)h1buf, (size_t)BATCH * HDIM / 2);
  zero_u32_kernel<<<1, 64, 0, stream>>>(bar, 64);

  // -------- one persistent kernel for the whole sequence + final FC
  persistent_lstm_kernel<<<NBLOCKS, 256, 0, stream>>>(
      xb, wxh0b, whh0b, wxh1b, whh1b, wfcb,
      bxh0, bhh0, bxh1, bhh1, bfc,
      c0, c1, h0buf, h1buf, out, bar);
}
